// WGAANLayer_43121471652249
// MI455X (gfx1250) — matrix operand order repaired and checked by reading, not compiled, not run on hardware
//
#include <hip/hip_runtime.h>
#include <stdint.h>

#define FEAT 64
#define TWO_F 128
#define BETA_W 0.5f
#define NEG_SLOPE 0.01f

typedef __attribute__((ext_vector_type(16))) __bf16 v16bf;
typedef __attribute__((ext_vector_type(8)))  float  v8f;

union BFrag {
  v16bf v;
  __bf16 b[16];
  unsigned short s[16];
  uint4 q[2];
};

// f32 -> bf16 bits, round-to-nearest-even (used only in one-time prep kernels)
__device__ __forceinline__ unsigned short f2bf(float x) {
  unsigned int u = __float_as_uint(x);
  unsigned int r = (u + 0x7FFFu + ((u >> 16) & 1u)) >> 16;
  return (unsigned short)r;
}

// ---------------------------------------------------------------------------
// K0: zero output + colsum, init min/max stats
// ---------------------------------------------------------------------------
__global__ void k_init(float* __restrict__ out, int out_n,
                       float* __restrict__ colsum, float* __restrict__ stats) {
  int gid = blockIdx.x * blockDim.x + threadIdx.x;
  int stride = gridDim.x * blockDim.x;
  for (int i = gid; i < out_n; i += stride) out[i] = 0.0f;
  if (gid < FEAT) colsum[gid] = 0.0f;
  if (gid == FEAT + 0) stats[0] = __uint_as_float(0x7F800000u);  // amin = +inf
  if (gid == FEAT + 1) stats[1] = __uint_as_float(0xFF800000u);  // amax = -inf
  if (gid == FEAT + 2) stats[2] = __uint_as_float(0x7F800000u);  // cmin
  if (gid == FEAT + 3) stats[3] = __uint_as_float(0xFF800000u);  // cmax
}

// ---------------------------------------------------------------------------
// K1: global min/max of amount & count (non-negative -> int-punned atomics)
// ---------------------------------------------------------------------------
__global__ void k_minmax(const float* __restrict__ amount,
                         const float* __restrict__ count,
                         int E, float* __restrict__ stats) {
  int gid = blockIdx.x * blockDim.x + threadIdx.x;
  int stride = gridDim.x * blockDim.x;
  float amin = __uint_as_float(0x7F800000u), amax = -amin;
  float cmin = amin, cmax = amax;
  for (int i = gid; i < E; i += stride) {
    float a = amount[i], c = count[i];
    amin = fminf(amin, a); amax = fmaxf(amax, a);
    cmin = fminf(cmin, c); cmax = fmaxf(cmax, c);
  }
#pragma unroll
  for (int m = 16; m >= 1; m >>= 1) {
    amin = fminf(amin, __shfl_xor(amin, m, 32));
    amax = fmaxf(amax, __shfl_xor(amax, m, 32));
    cmin = fminf(cmin, __shfl_xor(cmin, m, 32));
    cmax = fmaxf(cmax, __shfl_xor(cmax, m, 32));
  }
  if ((threadIdx.x & 31) == 0) {
    atomicMin((int*)&stats[0], __float_as_int(amin));
    atomicMax((int*)&stats[1], __float_as_int(amax));
    atomicMin((int*)&stats[2], __float_as_int(cmin));
    atomicMax((int*)&stats[3], __float_as_int(cmax));
  }
}

// ---------------------------------------------------------------------------
// K2a: fc_w (64 x 128 f32) -> bf16 bits in workspace
// ---------------------------------------------------------------------------
__global__ void k_cvt_w(const float* __restrict__ fc_w,
                        unsigned short* __restrict__ wbf) {
  int gid = blockIdx.x * blockDim.x + threadIdx.x;
  int stride = gridDim.x * blockDim.x;
  for (int i = gid; i < FEAT * TWO_F; i += stride) wbf[i] = f2bf(fc_w[i]);
}

// ---------------------------------------------------------------------------
// K2b: h (N x 64 f32) -> bf16 table in workspace (one-time; removes all
// conversion VALU from the hot GEMM loops and halves gather bytes)
// ---------------------------------------------------------------------------
__global__ void k_cvt_h(const float* __restrict__ h,
                        unsigned short* __restrict__ hbf, int n_elems) {
  int gid = blockIdx.x * blockDim.x + threadIdx.x;
  int stride = gridDim.x * blockDim.x;
  // process float4 -> 4 bf16 (one b64 store) at a time
  int n4 = n_elems >> 2;
  const float4* h4 = (const float4*)h;
  for (int i = gid; i < n4; i += stride) {
    float4 f = h4[i];
    union { unsigned short s[4]; uint2 u; } o;
    o.s[0] = f2bf(f.x); o.s[1] = f2bf(f.y);
    o.s[2] = f2bf(f.z); o.s[3] = f2bf(f.w);
    ((uint2*)hbf)[i] = o.u;
  }
}

// ---------------------------------------------------------------------------
// B fragments: B[k][n] = fc_w[n][k]; lane = column n (mod 16), half-wave
// selects the K set (lanes 0-15: K {0..7,16..23} of chunk, lanes 16-31: K {8..15,24..31}).
// ---------------------------------------------------------------------------
__device__ __forceinline__ void load_B_frags(const unsigned short* __restrict__ wbf,
                                             int m, int koff16, v16bf B[4][4]) {
#pragma unroll
  for (int nt = 0; nt < 4; ++nt) {
    int n = nt * 16 + m;
#pragma unroll
    for (int kc = 0; kc < 4; ++kc) {
      BFrag u;
      const uint4* p = (const uint4*)(wbf + n * TWO_F + kc * 32 + (koff16 >> 1));
      u.q[0] = p[0];
      u.q[1] = p[2];
      B[nt][kc] = u.v;
    }
  }
}

// ---------------------------------------------------------------------------
// A fragments: row = lane%16 (edge), lanes<16 hold K {0..7,16..23} of each
// 32-chunk, lanes>=16 hold K {8..15,24..31}.  PRE=true: direct b128 loads
// from the bf16 table.  PRE=false: f32 loads + cast (fallback).
// ---------------------------------------------------------------------------
template <bool PRE>
__device__ __forceinline__ void gather_A(const float* __restrict__ rs_f,
                                         const float* __restrict__ rd_f,
                                         const unsigned short* __restrict__ rs_b,
                                         const unsigned short* __restrict__ rd_b,
                                         int koff, v16bf A[4]) {
#pragma unroll
  for (int kc = 0; kc < 4; ++kc) {
    const int fb = (kc & 1) * 32 + koff;
    BFrag u;
    if constexpr (PRE) {
      const unsigned short* r = (kc < 2) ? rs_b : rd_b;
      u.q[0] = *(const uint4*)(r + fb);
      u.q[1] = *(const uint4*)(r + fb + 16);
    } else {
      const float* r = (kc < 2) ? rs_f : rd_f;
      float4 x0 = *(const float4*)(r + fb);
      float4 x1 = *(const float4*)(r + fb + 4);
      float4 y0 = *(const float4*)(r + fb + 16);
      float4 y1 = *(const float4*)(r + fb + 20);
      u.b[0]  = (__bf16)x0.x; u.b[1]  = (__bf16)x0.y;
      u.b[2]  = (__bf16)x0.z; u.b[3]  = (__bf16)x0.w;
      u.b[4]  = (__bf16)x1.x; u.b[5]  = (__bf16)x1.y;
      u.b[6]  = (__bf16)x1.z; u.b[7]  = (__bf16)x1.w;
      u.b[8]  = (__bf16)y0.x; u.b[9]  = (__bf16)y0.y;
      u.b[10] = (__bf16)y0.z; u.b[11] = (__bf16)y0.w;
      u.b[12] = (__bf16)y1.x; u.b[13] = (__bf16)y1.y;
      u.b[14] = (__bf16)y1.z; u.b[15] = (__bf16)y1.w;
    }
    A[kc] = u.v;
  }
}

// ---------------------------------------------------------------------------
// K3: pass A — GEMM + leaky_relu + edge weight, per-column sum of exp(score).
// Register accumulation across the whole grid-stride loop; one atomic per
// column per wave at the end.
// ---------------------------------------------------------------------------
template <bool PRE>
__global__ void __launch_bounds__(256) k_colsum(
    const float* __restrict__ h, const unsigned short* __restrict__ hbf,
    const long long* __restrict__ adj,
    const float* __restrict__ amount, const float* __restrict__ count,
    const unsigned short* __restrict__ wbf, const float* __restrict__ fc_b,
    const float* __restrict__ stats, float* __restrict__ colsum,
    int E, int numTiles) {
  const int lane = threadIdx.x & 31;
  const int m = lane & 15;
  const bool hi = lane >= 16;
  const int hi8 = hi ? 8 : 0;
  const int koff = hi ? 8 : 0;
  const int koff16 = hi ? 16 : 0;
  const int waveGid = blockIdx.x * (blockDim.x >> 5) + (threadIdx.x >> 5);
  const int waveStride = gridDim.x * (blockDim.x >> 5);
  const long long* adj_src = adj;
  const long long* adj_dst = adj + E;

  const float amin = stats[0], amax = stats[1], cmin = stats[2], cmax = stats[3];
  const float ia = BETA_W / (amax - amin + 1e-8f);
  const float ic = (1.0f - BETA_W) / (cmax - cmin + 1e-8f);

  v16bf B[4][4];
  load_B_frags(wbf, m, koff16, B);
  float bias[4];
#pragma unroll
  for (int nt = 0; nt < 4; ++nt) bias[nt] = fc_b[nt * 16 + m];

  float lsum[4] = {0.f, 0.f, 0.f, 0.f};

  for (int tile = waveGid; tile < numTiles; tile += waveStride) {
    const int e0 = tile * 16;
    int ea = e0 + m;
    if (ea >= E) ea = E - 1;
    // own-edge values; rows are distributed across lanes via shuffles
    const float ewOwn =
        (amount[ea] - amin) * ia + (count[ea] - cmin) * ic;
    const long long sOwn = adj_src[ea];
    const long long dOwn = adj_dst[ea];

    const float* rs_f = h + (size_t)sOwn * FEAT;
    const float* rd_f = h + (size_t)dOwn * FEAT;
    const unsigned short* rs_b = hbf + (size_t)sOwn * FEAT;
    const unsigned short* rd_b = hbf + (size_t)dOwn * FEAT;
    v16bf A[4];
    gather_A<PRE>(rs_f, rd_f, rs_b, rd_b, koff, A);

    float ew8[8];
#pragma unroll
    for (int g = 0; g < 8; ++g) ew8[g] = __shfl(ewOwn, hi8 + g, 32);

#pragma unroll
    for (int nt = 0; nt < 4; ++nt) {
      v8f c = {};
#pragma unroll
      for (int kc = 0; kc < 4; ++kc)
        c = __builtin_amdgcn_wmma_f32_16x16x32_bf16(false, A[kc], false,
                                                    B[nt][kc], (short)0, c,
                                                    false, false);
#pragma unroll
      for (int g = 0; g < 8; ++g) {
        float s = c[g] + bias[nt];
        s = (s >= 0.0f) ? s : NEG_SLOPE * s;
        s *= ew8[g];
        lsum[nt] += __expf(s);  // |s| bounded -> unshifted softmax is safe
      }
    }
  }
#pragma unroll
  for (int nt = 0; nt < 4; ++nt) {
    float v = lsum[nt] + __shfl_xor(lsum[nt], 16, 32);
    if (!hi)
      __hip_atomic_fetch_add(&colsum[nt * 16 + m], v, __ATOMIC_RELAXED,
                             __HIP_MEMORY_SCOPE_AGENT);
  }
}

// ---------------------------------------------------------------------------
// K4: pass B — recompute GEMM, attention = exp(s)/colsum, scatter-add
// attention * h_j into out[src] (output region is L2-resident).
// ---------------------------------------------------------------------------
template <bool PRE>
__global__ void __launch_bounds__(256) k_scatter(
    const float* __restrict__ h, const unsigned short* __restrict__ hbf,
    const long long* __restrict__ adj,
    const float* __restrict__ amount, const float* __restrict__ count,
    const unsigned short* __restrict__ wbf, const float* __restrict__ fc_b,
    const float* __restrict__ stats, const float* __restrict__ colsum,
    float* __restrict__ out, int E, int numTiles) {
  const int lane = threadIdx.x & 31;
  const int m = lane & 15;
  const bool hi = lane >= 16;
  const int hi8 = hi ? 8 : 0;
  const int koff = hi ? 8 : 0;
  const int koff16 = hi ? 16 : 0;
  const int waveGid = blockIdx.x * (blockDim.x >> 5) + (threadIdx.x >> 5);
  const int waveStride = gridDim.x * (blockDim.x >> 5);
  const long long* adj_src = adj;
  const long long* adj_dst = adj + E;

  const float amin = stats[0], amax = stats[1], cmin = stats[2], cmax = stats[3];
  const float ia = BETA_W / (amax - amin + 1e-8f);
  const float ic = (1.0f - BETA_W) / (cmax - cmin + 1e-8f);

  v16bf B[4][4];
  load_B_frags(wbf, m, koff16, B);
  float bias[4], invsum[4];
#pragma unroll
  for (int nt = 0; nt < 4; ++nt) {
    bias[nt] = fc_b[nt * 16 + m];
    invsum[nt] = 1.0f / colsum[nt * 16 + m];  // sums of exp() -> > 0
  }

  for (int tile = waveGid; tile < numTiles; tile += waveStride) {
    const int e0 = tile * 16;
    int ea = e0 + m;
    if (ea >= E) ea = E - 1;
    const float ewOwn =
        (amount[ea] - amin) * ia + (count[ea] - cmin) * ic;
    const long long sOwn = adj_src[ea];
    const long long dOwn = adj_dst[ea];

    const float* rs_f = h + (size_t)sOwn * FEAT;
    const float* rd_f = h + (size_t)dOwn * FEAT;
    const unsigned short* rs_b = hbf + (size_t)sOwn * FEAT;
    const unsigned short* rd_b = hbf + (size_t)dOwn * FEAT;
    v16bf A[4];
    gather_A<PRE>(rs_f, rd_f, rs_b, rd_b, koff, A);

    // distribute per-row edge weight and adjacency via shuffles (indices were
    // already loaded for the A gather — no redundant scalarized vmem)
    float ew8[8];
    long long sg[8], dg[8];
#pragma unroll
    for (int g = 0; g < 8; ++g) {
      ew8[g] = __shfl(ewOwn, hi8 + g, 32);
      sg[g] = __shfl(sOwn, hi8 + g, 32);
      dg[g] = __shfl(dOwn, hi8 + g, 32);
    }

#pragma unroll
    for (int nt = 0; nt < 4; ++nt) {
      v8f c = {};
#pragma unroll
      for (int kc = 0; kc < 4; ++kc)
        c = __builtin_amdgcn_wmma_f32_16x16x32_bf16(false, A[kc], false,
                                                    B[nt][kc], (short)0, c,
                                                    false, false);
      const int n = nt * 16 + m;
#pragma unroll
      for (int g = 0; g < 8; ++g) {
        float s = c[g] + bias[nt];
        s = (s >= 0.0f) ? s : NEG_SLOPE * s;
        s *= ew8[g];
        float att = __expf(s) * invsum[nt];
        float hj = h[(size_t)dg[g] * FEAT + n];
        __hip_atomic_fetch_add(&out[(size_t)sg[g] * FEAT + n], att * hj,
                               __ATOMIC_RELAXED, __HIP_MEMORY_SCOPE_AGENT);
      }
    }
  }
}

// ---------------------------------------------------------------------------
extern "C" void kernel_launch(void* const* d_in, const int* in_sizes, int n_in,
                              void* d_out, int out_size, void* d_ws,
                              size_t ws_size, hipStream_t stream) {
  const float* h = (const float*)d_in[0];
  const long long* adj = (const long long*)d_in[1];
  const float* amount = (const float*)d_in[2];
  const float* count = (const float*)d_in[3];
  const float* fc_w = (const float*)d_in[4];
  const float* fc_b = (const float*)d_in[5];
  float* out = (float*)d_out;

  const int E = in_sizes[2];
  const int numTiles = (E + 15) / 16;
  const int n_h = in_sizes[0];  // N * 64

  char* ws = (char*)d_ws;
  float* stats = (float*)ws;                            //    4 floats @ 0
  float* colsum = (float*)(ws + 64);                    //   64 floats @ 64
  unsigned short* wbf = (unsigned short*)(ws + 512);    // 8192 bf16   @ 512
  unsigned short* hbf = (unsigned short*)(ws + 32768);  // N*64 bf16   @ 32K

  const size_t need = 32768 + (size_t)n_h * sizeof(unsigned short);
  const bool pre = (ws_size >= need);

  k_init<<<4096, 256, 0, stream>>>(out, out_size, colsum, stats);
  k_minmax<<<2048, 256, 0, stream>>>(amount, count, E, stats);
  k_cvt_w<<<32, 256, 0, stream>>>(fc_w, wbf);
  if (pre) {
    k_cvt_h<<<4096, 256, 0, stream>>>(h, hbf, n_h);
    k_colsum<true><<<1024, 256, 0, stream>>>(h, hbf, adj, amount, count, wbf,
                                             fc_b, stats, colsum, E, numTiles);
    k_scatter<true><<<2048, 256, 0, stream>>>(h, hbf, adj, amount, count, wbf,
                                              fc_b, stats, colsum, out, E,
                                              numTiles);
  } else {
    k_colsum<false><<<1024, 256, 0, stream>>>(h, hbf, adj, amount, count, wbf,
                                              fc_b, stats, colsum, E, numTiles);
    k_scatter<false><<<2048, 256, 0, stream>>>(h, hbf, adj, amount, count, wbf,
                                               fc_b, stats, colsum, out, E,
                                               numTiles);
  }
}
